// Attention1D_48661979463991
// MI455X (gfx1250) — matrix operand order repaired and checked by reading, not compiled, not run on hardware
//
#include <hip/hip_runtime.h>

typedef __attribute__((ext_vector_type(16))) _Float16 v16h;
typedef __attribute__((ext_vector_type(8)))  _Float16 v8h;
typedef __attribute__((ext_vector_type(8)))  float    v8f;

constexpr int   Bb = 4;
constexpr int   Ll = 4096;
constexpr int   Cc = 64;
constexpr int   QSPLIT = 4;           // q-axis split for the stats pass
constexpr float SCALE = 0.125f;       // 1/sqrt(64)

// ---- fragment loaders (CDNA5 wave32 WMMA f16 16x16x32 layouts, ISA 7.12.2) ----
// A (16xK rows): lane m = lane%16, half = lane/16.
//   element j<8 : K = half*8 + j ; j>=8 : K = 16 + half*8 + (j-8)  -> two 16B loads
__device__ __forceinline__ v16h load_a_frag(const _Float16* rowbase, int half_) {
  v8h lo = *(const v8h*)(rowbase + half_ * 8);
  v8h hi = *(const v8h*)(rowbase + 16 + half_ * 8);
  v16h r;
#pragma unroll
  for (int j = 0; j < 8; ++j) { r[j] = lo[j]; r[j + 8] = hi[j]; }
  return r;
}
// B (Kx16 cols): lane n = lane%16, half = lane/16; element j<8 : K = half*8 + j ; j>=8 : K = 16 + half*8 + (j-8)
//   source is B^T stored row-major (row = n, K contiguous) -> two 16B loads
__device__ __forceinline__ v16h load_b_frag(const _Float16* rowbase, int half_) {
  v8h lo = *(const v8h*)(rowbase + half_ * 8);
  v8h hi = *(const v8h*)(rowbase + 16 + half_ * 8);
  v16h r;
#pragma unroll
  for (int j = 0; j < 8; ++j) { r[j] = lo[j]; r[j + 8] = hi[j]; }
  return r;
}

// ---------------- kernel 1: QKV projection (x @ W + b), f16 outputs ----------------
__global__ __launch_bounds__(64)
void qkv_kernel(const float* __restrict__ x,
                const float* __restrict__ Wq, const float* __restrict__ bq,
                const float* __restrict__ Wk, const float* __restrict__ bk,
                const float* __restrict__ Wv, const float* __restrict__ bv,
                _Float16* __restrict__ Qh, _Float16* __restrict__ Kh,
                _Float16* __restrict__ Vt) {
  __shared__ float xrow[Cc];
  const int row = blockIdx.x;     // 0 .. B*L-1
  const int f   = threadIdx.x;    // 0 .. 63
  xrow[f] = x[(size_t)row * Cc + f];
  __syncthreads();
  float q = bq[f], k = bk[f], v = bv[f];
#pragma unroll
  for (int c = 0; c < Cc; ++c) {
    const float xv = xrow[c];
    q += xv * Wq[c * Cc + f];
    k += xv * Wk[c * Cc + f];
    v += xv * Wv[c * Cc + f];
  }
  Qh[(size_t)row * Cc + f] = (_Float16)q;
  Kh[(size_t)row * Cc + f] = (_Float16)k;
  const int b = row / Ll;
  const int l = row % Ll;
  Vt[((size_t)b * Cc + f) * Ll + l] = (_Float16)v;   // V transposed: [b][f][l]
}

// ------- kernel 2: partial column-softmax stats over a q-slab (softmax is axis=1!) -------
// Each wave owns TWO 16-wide kv tiles (K frags loaded once, amortize every Q frag load).
// gridDim.z = QSPLIT slabs of q; partial (m,Z) combined by kernel 2b.
__global__ __launch_bounds__(128)
void colstats_kernel(const _Float16* __restrict__ Qh,
                     const _Float16* __restrict__ Kh,
                     float* __restrict__ Mp, float* __restrict__ Zp) {
  const int b     = blockIdx.y;
  const int qs    = blockIdx.z;
  const int lane  = threadIdx.x & 31;
  const int wave  = threadIdx.x >> 5;
  const int n     = lane & 15;
  const int half_ = lane >> 4;

  int  kvt[2];
  v16h bk[2][2];
#pragma unroll
  for (int t = 0; t < 2; ++t) {
    kvt[t] = blockIdx.x * 128 + (t * 4 + wave) * 16 + n;
    const _Float16* Krow = Kh + (size_t)(b * Ll + kvt[t]) * Cc;
    bk[t][0] = load_b_frag(Krow +  0, half_);
    bk[t][1] = load_b_frag(Krow + 32, half_);
  }

  float m_run[2] = {-3.0e38f, -3.0e38f};
  float z_run[2] = {0.0f, 0.0f};

  const int q_begin = qs * (Ll / QSPLIT);
  const int q_end   = q_begin + (Ll / QSPLIT);
  for (int q0 = q_begin; q0 < q_end; q0 += 16) {
    const _Float16* Qrow = Qh + (size_t)(b * Ll + q0 + n) * Cc;  // A row m = lane%16
    const v16h aq0 = load_a_frag(Qrow +  0, half_);
    const v16h aq1 = load_a_frag(Qrow + 32, half_);
#pragma unroll
    for (int t = 0; t < 2; ++t) {
      v8f acc = {};
      acc = __builtin_amdgcn_wmma_f32_16x16x32_f16(false, aq0, false, bk[t][0], (short)0, acc, false, false);
      acc = __builtin_amdgcn_wmma_f32_16x16x32_f16(false, aq1, false, bk[t][1], (short)0, acc, false, false);

      float cmax = -3.0e38f;
#pragma unroll
      for (int r = 0; r < 8; ++r) cmax = fmaxf(cmax, acc[r] * SCALE);
      cmax = fmaxf(cmax, __shfl_xor(cmax, 16, 32));   // combine lane halves (rows m+8)
      const float m_new = fmaxf(m_run[t], cmax);
      float psum = 0.0f;
#pragma unroll
      for (int r = 0; r < 8; ++r) psum += __expf(acc[r] * SCALE - m_new);
      psum += __shfl_xor(psum, 16, 32);
      z_run[t] = z_run[t] * __expf(m_run[t] - m_new) + psum;  // online rescale
      m_run[t] = m_new;
    }
  }
  if (lane < 16) {
#pragma unroll
    for (int t = 0; t < 2; ++t) {
      const size_t idx = (size_t)(qs * Bb + b) * Ll + kvt[t];
      Mp[idx] = m_run[t];
      Zp[idx] = z_run[t];
    }
  }
}

// ------- kernel 2b: combine per-slab partial stats -------
__global__ __launch_bounds__(256)
void combine_stats_kernel(const float* __restrict__ Mp, const float* __restrict__ Zp,
                          float* __restrict__ Mstat, float* __restrict__ Zstat) {
  const int i = blockIdx.x * 256 + threadIdx.x;   // 0 .. B*L-1
  float m = -3.0e38f;
#pragma unroll
  for (int s = 0; s < QSPLIT; ++s) m = fmaxf(m, Mp[(size_t)s * Bb * Ll + i]);
  float z = 0.0f;
#pragma unroll
  for (int s = 0; s < QSPLIT; ++s)
    z += Zp[(size_t)s * Bb * Ll + i] * __expf(Mp[(size_t)s * Bb * Ll + i] - m);
  Mstat[i] = m;
  Zstat[i] = z;
}

// ------- kernel 3: out = softmax_cols(Q K^T/8) @ V + x -------
// 64 q rows per block (4 persistent A-frag pairs / wave) => K,V read 4x less often.
__global__ __launch_bounds__(128)
void attn_out_kernel(const _Float16* __restrict__ Qh,
                     const _Float16* __restrict__ Kh,
                     const _Float16* __restrict__ Vt,
                     const float* __restrict__ Mstat,
                     const float* __restrict__ Zstat,
                     const float* __restrict__ x,
                     float* __restrict__ out) {
  __shared__ __align__(16) _Float16 Plds[64][72];  // 64 q-rows x 64 kv, padded stride
  const int b     = blockIdx.y;
  const int qblk  = blockIdx.x;         // 64 q rows per block
  const int lane  = threadIdx.x & 31;
  const int wave  = threadIdx.x >> 5;   // wave = f-tile (4 x 16 = 64 features)
  const int n     = lane & 15;
  const int half_ = lane >> 4;

  v16h aq[4][2];
#pragma unroll
  for (int t = 0; t < 4; ++t) {
    const _Float16* Qrow = Qh + (size_t)(b * Ll + qblk * 64 + t * 16 + n) * Cc;
    aq[t][0] = load_a_frag(Qrow +  0, half_);
    aq[t][1] = load_a_frag(Qrow + 32, half_);
  }
  v8f oacc[4] = {};

  for (int kv0 = 0; kv0 < Ll; kv0 += 64) {
    // --- each wave: 16-wide kv chunk, score tiles for all 4 q tiles ---
    const int kv = kv0 + wave * 16 + n;
    const _Float16* Krow = Kh + (size_t)(b * Ll + kv) * Cc;
    const v16h bk0 = load_b_frag(Krow +  0, half_);
    const v16h bk1 = load_b_frag(Krow + 32, half_);
    const float mk = Mstat[b * Ll + kv];
    const float rz = 1.0f / Zstat[b * Ll + kv];

#pragma unroll
    for (int t = 0; t < 4; ++t) {
      v8f sacc = {};
      sacc = __builtin_amdgcn_wmma_f32_16x16x32_f16(false, aq[t][0], false, bk0, (short)0, sacc, false, false);
      sacc = __builtin_amdgcn_wmma_f32_16x16x32_f16(false, aq[t][1], false, bk1, (short)0, sacc, false, false);
#pragma unroll
      for (int r = 0; r < 8; ++r) {
        const float p = __expf(sacc[r] * SCALE - mk) * rz;      // column-normalized prob
        Plds[t * 16 + r + 8 * half_][wave * 16 + n] = (_Float16)p;
      }
    }
    __syncthreads();

    // --- P(64q x 64kv) @ V(64kv x 16f) per wave f tile ---
#pragma unroll
    for (int kk = 0; kk < 2; ++kk) {
      const _Float16* Vrow = Vt + ((size_t)(b * Cc + wave * 16 + n)) * Ll + kv0 + kk * 32;
      const v16h bv = load_b_frag(Vrow, half_);
#pragma unroll
      for (int t = 0; t < 4; ++t) {
        const int row = t * 16 + n;
        v8h lo = *(const v8h*)&Plds[row][kk * 32 + half_ * 8];
        v8h hi = *(const v8h*)&Plds[row][kk * 32 + 16 + half_ * 8];
        v16h ap;
#pragma unroll
        for (int j = 0; j < 8; ++j) { ap[j] = lo[j]; ap[j + 8] = hi[j]; }
        oacc[t] = __builtin_amdgcn_wmma_f32_16x16x32_f16(false, ap, false, bv, (short)0, oacc[t], false, false);
      }
    }
    __syncthreads();
  }

  // epilogue: residual add, D layout scatter (m = r + 8*half, n = f)
#pragma unroll
  for (int t = 0; t < 4; ++t) {
#pragma unroll
    for (int r = 0; r < 8; ++r) {
      const int q = qblk * 64 + t * 16 + r + 8 * half_;
      const size_t idx = (size_t)(b * Ll + q) * Cc + wave * 16 + n;
      out[idx] = oacc[t][r] + x[idx];
    }
  }
}

extern "C" void kernel_launch(void* const* d_in, const int* in_sizes, int n_in,
                              void* d_out, int out_size, void* d_ws, size_t ws_size,
                              hipStream_t stream) {
  const float* x  = (const float*)d_in[0];
  const float* Wq = (const float*)d_in[1];
  const float* bq = (const float*)d_in[2];
  const float* Wk = (const float*)d_in[3];
  const float* bk = (const float*)d_in[4];
  const float* Wv = (const float*)d_in[5];
  const float* bv = (const float*)d_in[6];
  float* out = (float*)d_out;

  // ws layout: Qh 2MB | Kh 2MB | Vt 2MB | Mp 256KB | Zp 256KB | M 64KB | Z 64KB
  char* ws = (char*)d_ws;
  const size_t qkvBytes  = (size_t)Bb * Ll * Cc * sizeof(_Float16);   // 2 MiB each
  const size_t statElems = (size_t)Bb * Ll;
  _Float16* Qh = (_Float16*)(ws);
  _Float16* Kh = (_Float16*)(ws + qkvBytes);
  _Float16* Vt = (_Float16*)(ws + 2 * qkvBytes);
  float* Mp    = (float*)(ws + 3 * qkvBytes);
  float* Zp    = Mp + QSPLIT * statElems;
  float* Mstat = Zp + QSPLIT * statElems;
  float* Zstat = Mstat + statElems;

  qkv_kernel<<<Bb * Ll, Cc, 0, stream>>>(x, Wq, bq, Wk, bk, Wv, bv, Qh, Kh, Vt);
  colstats_kernel<<<dim3(Ll / 128, Bb, QSPLIT), 128, 0, stream>>>(Qh, Kh, Mp, Zp);
  combine_stats_kernel<<<(Bb * Ll) / 256, 256, 0, stream>>>(Mp, Zp, Mstat, Zstat);
  attn_out_kernel<<<dim3(Ll / 64, Bb), 128, 0, stream>>>(Qh, Kh, Vt, Mstat, Zstat, x, out);
}
